// GraphSAGEReasoner_4054449127923
// MI455X (gfx1250) — compile-verified, matrix operand order checked by reading, not run on hardware
//
#include <hip/hip_runtime.h>
#include <hip/hip_bf16.h>
#include <math.h>

// ---------------------------------------------------------------------------
// Problem constants (match reference)
// ---------------------------------------------------------------------------
#define EMB   256
#define STEP  512          // 2*EMB
#define BSZ   512          // batch of paths
#define PLEN  6            // entity positions per path
#define NB    64           // sampled neighbors
#define CNN   128
#define NPOOL 65           // NB + 1 rows per (b,p)

// ---------------------------------------------------------------------------
// WMMA types / helpers (gfx1250, wave32)
// ---------------------------------------------------------------------------
typedef __attribute__((ext_vector_type(16))) __bf16 bf16x16;
typedef __attribute__((ext_vector_type(8)))  float  f32x8;

union Frag32B {            // one 16x32 bf16 A/B fragment = 8 VGPRs = 32 bytes/lane
  bf16x16 v;
  uint4   q[2];
};

__device__ inline f32x8 wmma_bf16(bf16x16 a, bf16x16 b, f32x8 c) {
  // D = A(16x32) * B(32x16) + C(16x16 f32)
  return __builtin_amdgcn_wmma_f32_16x16x32_bf16(
      /*neg_a=*/false, a, /*neg_b=*/false, b,
      /*c_mod=*/(short)0, c, /*reuse_a=*/false, /*reuse_b=*/false);
}

__device__ inline unsigned short f2bf(float x) {
  unsigned u = __builtin_bit_cast(unsigned, x);
  unsigned r = u + 0x7FFFu + ((u >> 16) & 1u);   // round-to-nearest-even
  return (unsigned short)(r >> 16);
}

__device__ inline uint4 pack8(float4 a, float4 b) {
  uint4 o;
  o.x = (unsigned)f2bf(a.x) | ((unsigned)f2bf(a.y) << 16);
  o.y = (unsigned)f2bf(a.z) | ((unsigned)f2bf(a.w) << 16);
  o.z = (unsigned)f2bf(b.x) | ((unsigned)f2bf(b.y) << 16);
  o.w = (unsigned)f2bf(b.z) | ((unsigned)f2bf(b.w) << 16);
  return o;
}

// ---------------------------------------------------------------------------
// Kernel 0: transpose + convert weights to bf16 [n][k] layout
// ---------------------------------------------------------------------------
__global__ void gsr_convert_kernel(const float* __restrict__ Wp,
                                   const float* __restrict__ c2w,
                                   const float* __restrict__ c3w,
                                   unsigned short* __restrict__ WtPool,
                                   unsigned short* __restrict__ Wt2,
                                   unsigned short* __restrict__ Wt3) {
  int tid = blockIdx.x * blockDim.x + threadIdx.x;
  int stride = gridDim.x * blockDim.x;
  for (int i = tid; i < 512 * 512; i += stride) {       // WtPool[g][f] = Wp[f][g]
    int g = i >> 9, f = i & 511;
    WtPool[(size_t)g * 512 + f] = f2bf(Wp[(size_t)f * 512 + g]);
  }
  for (int i = tid; i < 128 * 1024; i += stride) {      // Wt2[o][k] = c2w[k][o]
    int o = i >> 10, k = i & 1023;
    Wt2[(size_t)o * 1024 + k] = f2bf(c2w[(size_t)k * 128 + o]);
  }
  for (int i = tid; i < 128 * 1536; i += stride) {      // Wt3[o][k] = c3w[k][o]
    int o = i / 1536, k = i % 1536;
    Wt3[(size_t)o * 1536 + k] = f2bf(c3w[(size_t)k * 128 + o]);
  }
}

// ---------------------------------------------------------------------------
// Kernel 1: gather + pool GEMM (65x512 @ 512x512) + bias + relu + row-max.
// One block (512 thr = 16 waves) per (b,p).  K is the OUTERMOST compute loop:
// per k-step we stage an 80x32 bf16 A slice in LDS, each wave loads its two
// B fragments once (nothing loop-invariant to hoist -> no spills) and issues
// 10 WMMAs into 5x2 persistent accumulators (80 VGPRs).
// ---------------------------------------------------------------------------
__global__ __launch_bounds__(512) void gsr_pool_kernel(
    const float* __restrict__ ent, const float* __restrict__ rel,
    const unsigned short* __restrict__ WtPool, const float* __restrict__ bpool,
    const int* __restrict__ path_ent, const int* __restrict__ path_rel,
    const int* __restrict__ nb_ent, const int* __restrict__ nb_rel,
    float* __restrict__ feats) {
  __shared__ unsigned short ldsA[80 * 32];    // 5 KB: 80 rows x 32 k (bf16)

  const int bp  = blockIdx.x;                 // b*PLEN + p
  const int tid = threadIdx.x;
  const int w   = tid >> 5;                   // wave id 0..15
  const int l   = tid & 31;                   // lane
  const int col = l & 15;                     // N within subtile / M row for A frag
  const int kb  = (l >> 4) * 8;               // k base per ISA 16-bit layout
  const int half = l >> 4;

  float bias[2];
  #pragma unroll
  for (int nt = 0; nt < 2; ++nt) bias[nt] = bpool[w * 32 + nt * 16 + col];

  f32x8 acc[5][2] = {};                       // persistent: 80 VGPRs

  for (int ks = 0; ks < 16; ++ks) {
    __syncthreads();                          // previous slice fully consumed
    // ---- stage A slice: rows 0..64 gathered, rows 65..79 zeroed ----
    for (int idx = tid; idx < 80 * 4; idx += 512) {
      int r = idx >> 2, chunk = idx & 3;
      int f0 = ks * 32 + chunk * 8;
      uint4 val;
      if (r < NPOOL) {
        int eid, rid;
        if (r == 0) { eid = path_ent[bp]; rid = path_rel[bp]; }
        else        { eid = nb_ent[(size_t)bp * NB + r - 1];
                      rid = nb_rel[(size_t)bp * NB + r - 1]; }
        const float* src = (f0 < EMB) ? (ent + (size_t)eid * EMB + f0)
                                      : (rel + (size_t)rid * EMB + (f0 - EMB));
        float4 a = *(const float4*)src;
        float4 b = *(const float4*)(src + 4);
        val = pack8(a, b);
      } else {
        val = make_uint4(0u, 0u, 0u, 0u);
      }
      *(uint4*)&ldsA[r * 32 + chunk * 8] = val;
    }
    __syncthreads();

    // ---- B fragments for this k-step (2 per wave, used once: no hoisting) ----
    Frag32B fb[2];
    #pragma unroll
    for (int nt = 0; nt < 2; ++nt) {
      const unsigned short* bq =
          WtPool + (size_t)(w * 32 + nt * 16 + col) * 512 + ks * 32 + kb;
      fb[nt].q[0] = *(const uint4*)bq;
      fb[nt].q[1] = *(const uint4*)(bq + 16);
    }

    // ---- 5 m-tiles x 2 n-tiles WMMAs ----
    #pragma unroll
    for (int mt = 0; mt < 5; ++mt) {
      Frag32B fa;
      const unsigned short* ap = &ldsA[(mt * 16 + (l & 15)) * 32 + kb];
      fa.q[0] = *(const uint4*)ap;
      fa.q[1] = *(const uint4*)(ap + 16);
      acc[mt][0] = wmma_bf16(fa.v, fb[0].v, acc[mt][0]);
      acc[mt][1] = wmma_bf16(fa.v, fb[1].v, acc[mt][1]);
    }
  }

  // ---- bias + relu + max over valid rows (relu folded: mx init 0) ----
  float mx[2] = {0.f, 0.f};
  #pragma unroll
  for (int mt = 0; mt < 5; ++mt) {
    #pragma unroll
    for (int nt = 0; nt < 2; ++nt) {
      #pragma unroll
      for (int i = 0; i < 8; ++i) {
        int row = mt * 16 + i + 8 * half;     // D row for vgpr i, this lane half
        float hv = acc[mt][nt][i] + bias[nt];
        if (row < NPOOL) mx[nt] = fmaxf(mx[nt], hv);
      }
    }
  }

  // combine lane halves (rows 0-7 vs 8-15 share column l&15)
  #pragma unroll
  for (int nt = 0; nt < 2; ++nt) {
    float other = __shfl_xor(mx[nt], 16, 32);
    float m = fmaxf(mx[nt], other);
    if (l < 16) feats[(size_t)bp * 512 + w * 32 + nt * 16 + l] = m;
  }
}

// ---------------------------------------------------------------------------
// Kernel 2: Conv1D (VALID) as WMMA GEMM.  M = BSZ*Tcnt rows, N = 128, K = KW*512.
// Row (b,t) of A = feats[b, t .. t+KW-1, :] which is contiguous in feats.
// One block computes 16 rows x 128 cols; wave w owns cols [w*16, w*16+16).
// ---------------------------------------------------------------------------
__global__ __launch_bounds__(256) void gsr_conv_kernel(
    const float* __restrict__ feats, const unsigned short* __restrict__ Wt,
    const float* __restrict__ bias, float* __restrict__ out,
    int Tcnt, int K) {
  __shared__ unsigned short ldsA[16 * 1536];  // 48 KB max (K=1536)

  const int tid = threadIdx.x;
  const int mt  = blockIdx.x;
  const int w   = tid >> 5;
  const int l   = tid & 31;
  const int col = l & 15;
  const int kb  = (l >> 4) * 8;
  const int n   = w * 16 + col;

  const int chunksPerRow = K >> 3;
  for (int idx = tid; idx < 16 * chunksPerRow; idx += 256) {
    int r = idx / chunksPerRow;
    int c = idx % chunksPerRow;
    int row = mt * 16 + r;
    int b = row / Tcnt, t = row % Tcnt;
    const float* src = feats + ((size_t)b * PLEN + t) * 512 + c * 8;
    float4 a  = *(const float4*)src;
    float4 b2 = *(const float4*)(src + 4);
    *(uint4*)&ldsA[r * K + c * 8] = pack8(a, b2);
  }
  __syncthreads();

  f32x8 acc = {};
  const unsigned short* arow = &ldsA[(l & 15) * K];
  const unsigned short* brow = Wt + (size_t)n * K;
  for (int ks = 0; ks < (K >> 5); ++ks) {
    Frag32B fa, fb;
    const unsigned short* ap = arow + ks * 32 + kb;
    fa.q[0] = *(const uint4*)ap;
    fa.q[1] = *(const uint4*)(ap + 16);
    const unsigned short* bq = brow + ks * 32 + kb;
    fb.q[0] = *(const uint4*)bq;
    fb.q[1] = *(const uint4*)(bq + 16);
    acc = wmma_bf16(fa.v, fb.v, acc);
  }

  float bn = bias[n];
  #pragma unroll
  for (int i = 0; i < 8; ++i) {
    int m = i + 8 * (l >> 4);
    float v = acc[i] + bn;
    out[(size_t)(mt * 16 + m) * CNN + n] = v > 0.f ? v : 0.f;
  }
}

// ---------------------------------------------------------------------------
// Kernel 3: time max-pool + MLP(256->400->400->2) + softmax.  One block per b.
// ---------------------------------------------------------------------------
__global__ __launch_bounds__(256) void gsr_head_kernel(
    const float* __restrict__ c2, const float* __restrict__ c3,
    const float* __restrict__ d1w, const float* __restrict__ d1b,
    const float* __restrict__ d2w, const float* __restrict__ d2b,
    const float* __restrict__ d3w, const float* __restrict__ d3b,
    float* __restrict__ out) {
  __shared__ float x[256];
  __shared__ float y1[400];
  __shared__ float y2[400];
  const int b = blockIdx.x, tid = threadIdx.x;

  if (tid < 128) {
    float m = c2[((size_t)b * 5) * CNN + tid];
    for (int t = 1; t < 5; ++t) m = fmaxf(m, c2[((size_t)b * 5 + t) * CNN + tid]);
    x[tid] = m;
  } else {
    int o = tid - 128;
    float m = c3[((size_t)b * 4) * CNN + o];
    for (int t = 1; t < 4; ++t) m = fmaxf(m, c3[((size_t)b * 4 + t) * CNN + o]);
    x[tid] = m;
  }
  __syncthreads();

  for (int o = tid; o < 400; o += 256) {
    float s = d1b[o];
    for (int i = 0; i < 256; ++i) s += x[i] * d1w[(size_t)i * 400 + o];
    y1[o] = s > 0.f ? s : 0.f;
  }
  __syncthreads();

  for (int o = tid; o < 400; o += 256) {
    float s = d2b[o];
    for (int i = 0; i < 400; ++i) s += y1[i] * d2w[(size_t)i * 400 + o];
    y2[o] = s > 0.f ? s : 0.f;
  }
  __syncthreads();

  if (tid == 0) {
    float l0 = d3b[0], l1 = d3b[1];
    for (int i = 0; i < 400; ++i) {
      l0 += y2[i] * d3w[(size_t)i * 2 + 0];
      l1 += y2[i] * d3w[(size_t)i * 2 + 1];
    }
    float m = fmaxf(l0, l1);
    float e0 = expf(l0 - m), e1 = expf(l1 - m);
    float inv = 1.f / (e0 + e1);
    out[(size_t)b * 2 + 0] = e0 * inv;
    out[(size_t)b * 2 + 1] = e1 * inv;
  }
}

// ---------------------------------------------------------------------------
// Launch
// ---------------------------------------------------------------------------
extern "C" void kernel_launch(void* const* d_in, const int* in_sizes, int n_in,
                              void* d_out, int out_size, void* d_ws, size_t ws_size,
                              hipStream_t stream) {
  const float* ent_emb = (const float*)d_in[0];
  const float* rel_emb = (const float*)d_in[1];
  const float* W_pool  = (const float*)d_in[2];
  const float* b_pool  = (const float*)d_in[3];
  const float* conv2_w = (const float*)d_in[4];
  const float* conv2_b = (const float*)d_in[5];
  const float* conv3_w = (const float*)d_in[6];
  const float* conv3_b = (const float*)d_in[7];
  const float* d1_w    = (const float*)d_in[8];
  const float* d1_b    = (const float*)d_in[9];
  const float* d2_w    = (const float*)d_in[10];
  const float* d2_b    = (const float*)d_in[11];
  const float* d3_w    = (const float*)d_in[12];
  const float* d3_b    = (const float*)d_in[13];
  const int* path_ent  = (const int*)d_in[14];
  const int* path_rel  = (const int*)d_in[15];
  const int* nb_ent    = (const int*)d_in[16];
  const int* nb_rel    = (const int*)d_in[17];
  float* out = (float*)d_out;

  // workspace layout (bytes, 256B-aligned slots)
  char* ws = (char*)d_ws;
  unsigned short* WtPool = (unsigned short*)(ws + 0);         // 512*512*2  = 512 KB
  unsigned short* Wt2    = (unsigned short*)(ws + 524288);    // 128*1024*2 = 256 KB
  unsigned short* Wt3    = (unsigned short*)(ws + 786432);    // 128*1536*2 = 384 KB
  float* feats           = (float*)(ws + 1179648);            // 512*6*512*4 ≈ 6.3 MB
  float* c2              = (float*)(ws + 7471104);            // 512*5*128*4
  float* c3              = (float*)(ws + 8781824);            // 512*4*128*4
  // total ≈ 9.83 MB

  gsr_convert_kernel<<<512, 256, 0, stream>>>(W_pool, conv2_w, conv3_w,
                                              WtPool, Wt2, Wt3);

  gsr_pool_kernel<<<BSZ * PLEN, 512, 0, stream>>>(
      ent_emb, rel_emb, WtPool, b_pool,
      path_ent, path_rel, nb_ent, nb_rel, feats);

  gsr_conv_kernel<<<(BSZ * 5) / 16, 256, 0, stream>>>(feats, Wt2, conv2_b, c2,
                                                      /*Tcnt=*/5, /*K=*/1024);
  gsr_conv_kernel<<<(BSZ * 4) / 16, 256, 0, stream>>>(feats, Wt3, conv3_b, c3,
                                                      /*Tcnt=*/4, /*K=*/1536);

  gsr_head_kernel<<<BSZ, 256, 0, stream>>>(c2, c3, d1_w, d1_b, d2_w, d2_b,
                                           d3_w, d3_b, out);
}
